// MultiScaleGNN_86251533238778
// MI455X (gfx1250) — compile-verified, matrix operand order checked by reading, not run on hardware
//
#include <hip/hip_runtime.h>

// ---------------------------------------------------------------------------
// MultiScaleGNN for MI455X (gfx1250): all dense GEMMs via v_wmma_f32_16x16x32_bf16
// (16x64 output tile per wave32, fully unrolled K), graph scatters via
// global float atomics.
// ---------------------------------------------------------------------------

typedef __attribute__((ext_vector_type(16))) __bf16 bf16x16;
typedef __attribute__((ext_vector_type(8)))  float  v8f;

__device__ __forceinline__ __bf16 f2bf(float f) {
  unsigned int u = __float_as_uint(f);
  unsigned short s = (unsigned short)(u >> 16);   // truncate-to-bf16 (free: half-reg move)
  __bf16 r;
  __builtin_memcpy(&r, &s, 2);
  return r;
}

// Load one 16-element bf16 fragment for this lane: two contiguous 8-float
// chunks at k0 and k1 (matches the 16-bit A/B VGPR layout for 16x16x32).
__device__ __forceinline__ bf16x16 load_frag(const float* __restrict__ p, int k0, int k1) {
  float4 c0 = *(const float4*)(p + k0);
  float4 c1 = *(const float4*)(p + k0 + 4);
  float4 c2 = *(const float4*)(p + k1);
  float4 c3 = *(const float4*)(p + k1 + 4);
  bf16x16 v;
  v[0] = f2bf(c0.x); v[1] = f2bf(c0.y); v[2]  = f2bf(c0.z); v[3]  = f2bf(c0.w);
  v[4] = f2bf(c1.x); v[5] = f2bf(c1.y); v[6]  = f2bf(c1.z); v[7]  = f2bf(c1.w);
  v[8] = f2bf(c2.x); v[9] = f2bf(c2.y); v[10] = f2bf(c2.z); v[11] = f2bf(c2.w);
  v[12]= f2bf(c3.x); v[13]= f2bf(c3.y); v[14] = f2bf(c3.z); v[15] = f2bf(c3.w);
  return v;
}

__device__ __forceinline__ void atomicMaxF(float* addr, float val) {
  unsigned int* ua = reinterpret_cast<unsigned int*>(addr);
  unsigned int old = *ua;
  while (true) {
    float f = __uint_as_float(old);
    if (f >= val) break;
    unsigned int assumed = old;
    old = atomicCAS(ua, assumed, __float_as_uint(val));
    if (old == assumed) break;
  }
}

// ---------------------------------------------------------------------------
// Y[M,Nout] = A[M,K] @ W[Nout,K]^T  (+bias) (+Cin) (optional relu)
// One wave32 per 16(M) x 64(N) output tile: A fragment reused across 4
// independent WMMAs per K-chunk; K compile-time -> fully unrolled.
// ---------------------------------------------------------------------------
__device__ __forceinline__ void gemm_epilogue(const v8f& acc, const float* bias,
                                              const float* Cin, float* Y,
                                              int M, int Nout, int relu,
                                              int rowBase, int half, int ncol) {
  float bval = bias ? bias[ncol] : 0.0f;
#pragma unroll
  for (int r = 0; r < 8; ++r) {
    int row = rowBase + half * 8 + r;   // VGPR r: M=r (lanes 0-15) / M=r+8 (lanes 16-31)
    if (row < M) {
      float v = acc[r] + bval;
      if (Cin) v += Cin[(size_t)row * Nout + ncol];
      if (relu) v = fmaxf(v, 0.0f);
      Y[(size_t)row * Nout + ncol] = v;
    }
  }
}

template <int K>
__global__ void __launch_bounds__(256)
gemm_bf16_wmma_kernel(const float* __restrict__ A, const float* __restrict__ W,
                      const float* __restrict__ bias, const float* __restrict__ Cin,
                      float* __restrict__ Y, int M, int Nout, int relu) {
  int gwave = (blockIdx.x * blockDim.x + threadIdx.x) >> 5;
  int lane  = threadIdx.x & 31;
  int nt = Nout >> 6;              // 64-wide N tiles (all Nout here: 128/384/512)
  int mt = (M + 15) >> 4;
  if (gwave >= mt * nt) return;
  int tileM = gwave / nt;
  int tileN = gwave - tileM * nt;
  int half = lane >> 4;            // 0: lanes 0-15, 1: lanes 16-31
  int lidx = lane & 15;

  int arow = tileM * 16 + lidx; if (arow >= M) arow = M - 1;  // clamp; stores masked
  const float* ap = A + (size_t)arow * K;
  const float* wp = W + (size_t)(tileN * 64 + lidx) * K;      // B rows: +0/+16K/+32K/+48K

  v8f acc0 = {}, acc1 = {}, acc2 = {}, acc3 = {};
#pragma unroll
  for (int kk = 0; kk < K; kk += 32) {
    int k0 = kk + half * 8;        // frag elems 0..7  : K = k0..k0+7
    int k1 = kk + 16 + half * 8;   // frag elems 8..15 : K = k1..k1+7
    bf16x16 af = load_frag(ap, k0, k1);
    bf16x16 b0 = load_frag(wp,            k0, k1);
    bf16x16 b1 = load_frag(wp + 16 * K,   k0, k1);
    bf16x16 b2 = load_frag(wp + 32 * K,   k0, k1);
    bf16x16 b3 = load_frag(wp + 48 * K,   k0, k1);
    acc0 = __builtin_amdgcn_wmma_f32_16x16x32_bf16(false, af, false, b0, (short)0, acc0, false, false);
    acc1 = __builtin_amdgcn_wmma_f32_16x16x32_bf16(false, af, false, b1, (short)0, acc1, false, false);
    acc2 = __builtin_amdgcn_wmma_f32_16x16x32_bf16(false, af, false, b2, (short)0, acc2, false, false);
    acc3 = __builtin_amdgcn_wmma_f32_16x16x32_bf16(false, af, false, b3, (short)0, acc3, false, false);
  }

  int rowBase = tileM * 16;
  int ncol0 = tileN * 64 + lidx;
  gemm_epilogue(acc0, bias, Cin, Y, M, Nout, relu, rowBase, half, ncol0);
  gemm_epilogue(acc1, bias, Cin, Y, M, Nout, relu, rowBase, half, ncol0 + 16);
  gemm_epilogue(acc2, bias, Cin, Y, M, Nout, relu, rowBase, half, ncol0 + 32);
  gemm_epilogue(acc3, bias, Cin, Y, M, Nout, relu, rowBase, half, ncol0 + 48);
}

// ---------------------------------------------------------------------------
// Elementwise / scatter kernels (H == 128 throughout)
// ---------------------------------------------------------------------------
__global__ void fill_kernel(float* p, float v, long n) {
  long i = (long)blockIdx.x * blockDim.x + threadIdx.x;
  if (i < n) p[i] = v;
}

__global__ void scatter_add_kernel(const float* __restrict__ x, const int* __restrict__ src,
                                   const int* __restrict__ dst, float* __restrict__ agg, int E) {
  int t = blockIdx.x * blockDim.x + threadIdx.x;
  if (t >= E * 128) return;
  int e = t >> 7, f = t & 127;
  atomicAdd(&agg[(size_t)dst[e] * 128 + f], x[(size_t)src[e] * 128 + f]);
}

__global__ void count_kernel(const int* __restrict__ dst, float* __restrict__ cnt, int E) {
  int e = blockIdx.x * blockDim.x + threadIdx.x;
  if (e < E) atomicAdd(&cnt[dst[e]], 1.0f);
}

__global__ void sage_mean_kernel(const float* __restrict__ s, const float* __restrict__ cnt,
                                 float* __restrict__ mean, int n) {
  int idx = blockIdx.x * blockDim.x + threadIdx.x;
  if (idx >= n * 128) return;
  int i = idx >> 7;
  mean[idx] = s[idx] / fmaxf(cnt[i], 1.0f);
}

__global__ void pool_kernel(const float* __restrict__ h, float* __restrict__ out, int nOut, int P) {
  int idx = blockIdx.x * blockDim.x + threadIdx.x;
  if (idx >= nOut * 128) return;
  int i = idx >> 7, f = idx & 127;
  float s = 0.0f;
  for (int j = 0; j < P; ++j) s += h[(size_t)(i * P + j) * 128 + f];
  out[idx] = s / (float)P;
}

// ------------------------------ GAT -----------------------------------------
__global__ void gat_coeff_kernel(const float* __restrict__ hgat, const float* __restrict__ asrc,
                                 const float* __restrict__ adst, float* __restrict__ a_s,
                                 float* __restrict__ a_d, int n4) {
  int t = blockIdx.x * blockDim.x + threadIdx.x;
  if (t >= n4 * 4) return;
  int n = t >> 2, hh = t & 3;
  const float* hp = hgat + (size_t)n * 512 + hh * 128;
  const float* as = asrc + hh * 128;
  const float* ad = adst + hh * 128;
  float s = 0.0f, d = 0.0f;
  for (int c = 0; c < 128; ++c) { float v = hp[c]; s += v * as[c]; d += v * ad[c]; }
  a_s[t] = s; a_d[t] = d;
}

__device__ __forceinline__ void gat_edge_sd(const int* cei, int E2, int e, int& s, int& d) {
  if (e < E2) { s = cei[e]; d = cei[E2 + e]; } else { s = d = e - E2; }  // self loops appended
}

__global__ void gat_edge_max_kernel(const int* __restrict__ cei, int E2, int n4,
                                    const float* __restrict__ a_s, const float* __restrict__ a_d,
                                    float* __restrict__ ebuf, float* __restrict__ mbuf) {
  int ET = E2 + n4;
  int t = blockIdx.x * blockDim.x + threadIdx.x;
  if (t >= ET * 4) return;
  int e = t >> 2, hh = t & 3, s, d;
  gat_edge_sd(cei, E2, e, s, d);
  float v = a_s[s * 4 + hh] + a_d[d * 4 + hh];
  v = v > 0.0f ? v : 0.2f * v;                 // leaky relu
  ebuf[t] = v;
  atomicMaxF(&mbuf[d * 4 + hh], v);
}

__global__ void gat_edge_exp_kernel(const int* __restrict__ cei, int E2, int n4,
                                    float* __restrict__ ebuf, const float* __restrict__ mbuf,
                                    float* __restrict__ denom) {
  int ET = E2 + n4;
  int t = blockIdx.x * blockDim.x + threadIdx.x;
  if (t >= ET * 4) return;
  int e = t >> 2, hh = t & 3, s, d;
  gat_edge_sd(cei, E2, e, s, d);
  float ex = __expf(ebuf[t] - mbuf[d * 4 + hh]);
  ebuf[t] = ex;
  atomicAdd(&denom[d * 4 + hh], ex);
}

__global__ void gat_edge_aggr_kernel(const int* __restrict__ cei, int E2, int n4,
                                     const float* __restrict__ ebuf, const float* __restrict__ denom,
                                     const float* __restrict__ hgat, float* __restrict__ gout) {
  int ET = E2 + n4;
  int t = blockIdx.x * blockDim.x + threadIdx.x;
  if (t >= ET * 4) return;
  int e = t >> 2, hh = t & 3, s, d;
  gat_edge_sd(cei, E2, e, s, d);
  float alpha = ebuf[t] / denom[d * 4 + hh];
  const float* hp = hgat + (size_t)s * 512 + hh * 128;
  float* op = gout + (size_t)d * 512 + hh * 128;
  for (int c = 0; c < 128; ++c) atomicAdd(&op[c], alpha * hp[c]);
}

__global__ void gat_finalize_kernel(const float* __restrict__ gout, const float* __restrict__ b,
                                    float* __restrict__ out, int n4) {
  int idx = blockIdx.x * blockDim.x + threadIdx.x;
  if (idx >= n4 * 128) return;
  int n = idx >> 7, c = idx & 127;
  const float* gp = gout + (size_t)n * 512;
  float v = 0.25f * (gp[c] + gp[128 + c] + gp[256 + c] + gp[384 + c]) + b[c];
  out[idx] = fmaxf(v, 0.0f);
}

// ------------------------------ MHA / LN ------------------------------------
__global__ void stack_kernel(const float* __restrict__ x0, const float* __restrict__ x1,
                             const float* __restrict__ x2, float* __restrict__ st, int n) {
  int idx = blockIdx.x * blockDim.x + threadIdx.x;
  if (idx >= n * 128) return;
  int i = idx >> 7, f = idx & 127;
  size_t b = (size_t)i * 3 * 128 + f;
  st[b]       = x0[idx];
  st[b + 128] = x1[((i >> 1) << 7) + f];
  st[b + 256] = x2[((i >> 2) << 7) + f];
}

// per (node, head): 3x3 attention over L=3 tokens, dh=32; fold mean over L.
__global__ void mha_attn_kernel(const float* __restrict__ qkv, float* __restrict__ obar, int n_nodes) {
  int t = blockIdx.x * blockDim.x + threadIdx.x;
  if (t >= n_nodes * 4) return;
  int n = t >> 2, hh = t & 3;
  const float* base = qkv + (size_t)n * 3 * 384 + hh * 32;  // q:+0, k:+128, v:+256 per token
  float attn[3][3];
  for (int l = 0; l < 3; ++l) {
    float mx = -1e30f;
    for (int m = 0; m < 3; ++m) {
      float dot = 0.0f;
      for (int d = 0; d < 32; ++d) dot += base[l * 384 + d] * base[m * 384 + 128 + d];
      dot *= 0.17677669529663687f;  // 1/sqrt(32)
      attn[l][m] = dot;
      mx = fmaxf(mx, dot);
    }
    float ssum = 0.0f;
    for (int m = 0; m < 3; ++m) { attn[l][m] = __expf(attn[l][m] - mx); ssum += attn[l][m]; }
    float inv = 1.0f / ssum;
    for (int m = 0; m < 3; ++m) attn[l][m] *= inv;
  }
  for (int d = 0; d < 32; ++d) {
    float acc = 0.0f;
    for (int l = 0; l < 3; ++l)
      for (int m = 0; m < 3; ++m)
        acc += attn[l][m] * base[m * 384 + 256 + d];
    obar[(size_t)n * 128 + hh * 32 + d] = acc * (1.0f / 3.0f);
  }
}

__global__ void ln_kernel(const float* __restrict__ c, const float* __restrict__ g,
                          const float* __restrict__ b, float* __restrict__ out, int n) {
  int i = blockIdx.x * blockDim.x + threadIdx.x;
  if (i >= n) return;
  const float* cp = c + (size_t)i * 128;
  float sum = 0.0f, ss = 0.0f;
  for (int f = 0; f < 128; ++f) { float v = cp[f]; sum += v; ss += v * v; }
  float mu = sum * (1.0f / 128.0f);
  float var = ss * (1.0f / 128.0f) - mu * mu;
  float inv = rsqrtf(var + 1e-5f);
  float* op = out + (size_t)i * 128;
  for (int f = 0; f < 128; ++f) op[f] = (cp[f] - mu) * inv * g[f] + b[f];
}

// ---------------------------------------------------------------------------
// Host orchestration
// ---------------------------------------------------------------------------
static inline int gridFor(long n, int b) { return (int)((n + b - 1) / b); }

static inline void launch_gemm(const float* A, const float* W, const float* bias,
                               const float* Cin, float* Y, int M, int Nout, int K,
                               int relu, hipStream_t s) {
  long waves = (long)((M + 15) / 16) * (Nout / 64);
  int blocks = gridFor(waves * 32, 256);
  if (K == 64) {
    gemm_bf16_wmma_kernel<64><<<blocks, 256, 0, s>>>(A, W, bias, Cin, Y, M, Nout, relu);
  } else {
    gemm_bf16_wmma_kernel<128><<<blocks, 256, 0, s>>>(A, W, bias, Cin, Y, M, Nout, relu);
  }
}

static inline void launch_fill(float* p, float v, long n, hipStream_t s) {
  fill_kernel<<<gridFor(n, 256), 256, 0, s>>>(p, v, n);
}

extern "C" void kernel_launch(void* const* d_in, const int* in_sizes, int n_in,
                              void* d_out, int out_size, void* d_ws, size_t ws_size,
                              hipStream_t stream) {
  (void)n_in; (void)out_size; (void)ws_size;

  const float* x        = (const float*)d_in[0];
  const int*   ei       = (const int*)d_in[1];
  const int*   cei1     = (const int*)d_in[2];
  const int*   cei2     = (const int*)d_in[3];
  const float* node_w   = (const float*)d_in[4];
  const float* node_b   = (const float*)d_in[5];
  const float* gc_wrel  = (const float*)d_in[6];
  const float* gc_brel  = (const float*)d_in[7];
  const float* gc_wroot = (const float*)d_in[8];
  const float* sage_wl  = (const float*)d_in[9];
  const float* sage_bl  = (const float*)d_in[10];
  const float* sage_wr  = (const float*)d_in[11];
  const float* gat_w    = (const float*)d_in[12];
  const float* gat_asrc = (const float*)d_in[13];
  const float* gat_adst = (const float*)d_in[14];
  const float* gat_b    = (const float*)d_in[15];
  const float* mha_in_w = (const float*)d_in[16];
  const float* mha_in_b = (const float*)d_in[17];
  const float* mha_out_w= (const float*)d_in[18];
  const float* mha_out_b= (const float*)d_in[19];
  const float* ln_g     = (const float*)d_in[20];
  const float* ln_b     = (const float*)d_in[21];
  const float* fp_w     = (const float*)d_in[22];
  const float* fp_b     = (const float*)d_in[23];

  const int N  = in_sizes[0] / 64;   // 20000
  const int N2 = N / 2, N4 = N / 4;
  const int E  = in_sizes[1] / 2;
  const int E1 = in_sizes[2] / 2;
  const int E2 = in_sizes[3] / 2;
  const int ET = E2 + N4;

  float* ws = (float*)d_ws;
  size_t off = 0;
  auto alloc = [&](size_t n) -> float* {
    float* p = ws + off;
    off += (n + 31) & ~(size_t)31;   // keep 128B alignment for float4 loads
    return p;
  };

  float* h     = alloc((size_t)N * 128);
  float* x0a   = alloc((size_t)N * 128);
  float* x0b   = alloc((size_t)N * 128);
  float* agg   = alloc((size_t)N * 128);
  float* tmp   = alloc((size_t)N * 128);
  float* x1a   = alloc((size_t)N2 * 128);
  float* x1b   = alloc((size_t)N2 * 128);
  float* s1    = alloc((size_t)N2 * 128);
  float* cnt1  = alloc((size_t)N2);
  float* mean1 = alloc((size_t)N2 * 128);
  float* x2a   = alloc((size_t)N4 * 128);
  float* x2b   = alloc((size_t)N4 * 128);
  float* hgat  = alloc((size_t)N4 * 512);
  float* gout  = alloc((size_t)N4 * 512);
  float* a_s   = alloc((size_t)N4 * 4);
  float* a_d   = alloc((size_t)N4 * 4);
  float* mgat  = alloc((size_t)N4 * 4);
  float* dnm   = alloc((size_t)N4 * 4);
  float* ebuf  = alloc((size_t)ET * 4);
  float* stck  = alloc((size_t)N * 3 * 128);
  float* qkv   = alloc((size_t)N * 3 * 384);
  float* obar  = alloc((size_t)N * 128);
  float* comb  = alloc((size_t)N * 128);
  float* nrm   = alloc((size_t)N * 128);

  // ---- node embedding: h = x @ node_w^T + node_b  [N,64]x[128,64]^T --------
  launch_gemm(x, node_w, node_b, nullptr, h, N, 128, 64, 0, stream);

  // ---- fine scale: 2x GraphConv ------------------------------------------
  {
    const float* chain[3] = {h, x0a, x0b};
    for (int l = 0; l < 2; ++l) {
      const float* in = chain[l];
      float* outb = (float*)chain[l + 1];
      launch_fill(agg, 0.0f, (long)N * 128, stream);
      scatter_add_kernel<<<gridFor((long)E * 128, 256), 256, 0, stream>>>(in, ei, ei + E, agg, E);
      launch_gemm(in,  gc_wroot + l * 128 * 128, nullptr, nullptr, tmp, N, 128, 128, 0, stream);
      launch_gemm(agg, gc_wrel  + l * 128 * 128, gc_brel + l * 128, tmp, outb, N, 128, 128, 1, stream);
    }
  }

  // ---- mid scale: pool/2 + 2x SAGE ---------------------------------------
  pool_kernel<<<gridFor((long)N2 * 128, 256), 256, 0, stream>>>(h, x1a, N2, 2);
  {
    const float* chain[3] = {x1a, x1b, x1a};
    for (int l = 0; l < 2; ++l) {
      const float* in = chain[l];
      float* outb = (float*)chain[l + 1];
      launch_fill(s1, 0.0f, (long)N2 * 128, stream);
      launch_fill(cnt1, 0.0f, (long)N2, stream);
      scatter_add_kernel<<<gridFor((long)E1 * 128, 256), 256, 0, stream>>>(in, cei1, cei1 + E1, s1, E1);
      count_kernel<<<gridFor((long)E1, 256), 256, 0, stream>>>(cei1 + E1, cnt1, E1);
      sage_mean_kernel<<<gridFor((long)N2 * 128, 256), 256, 0, stream>>>(s1, cnt1, mean1, N2);
      launch_gemm(in,    sage_wr + l * 128 * 128, nullptr, nullptr, tmp, N2, 128, 128, 0, stream);
      launch_gemm(mean1, sage_wl + l * 128 * 128, sage_bl + l * 128, tmp, outb, N2, 128, 128, 1, stream);
    }
  }

  // ---- coarse scale: pool/4 + 2x GAT -------------------------------------
  pool_kernel<<<gridFor((long)N4 * 128, 256), 256, 0, stream>>>(h, x2a, N4, 4);
  {
    const float* chain[3] = {x2a, x2b, x2a};
    for (int l = 0; l < 2; ++l) {
      const float* in = chain[l];
      float* outb = (float*)chain[l + 1];
      launch_gemm(in, gat_w + l * 512 * 128, nullptr, nullptr, hgat, N4, 512, 128, 0, stream);
      gat_coeff_kernel<<<gridFor((long)N4 * 4, 256), 256, 0, stream>>>(
          hgat, gat_asrc + l * 4 * 128, gat_adst + l * 4 * 128, a_s, a_d, N4);
      launch_fill(mgat, -1e30f, (long)N4 * 4, stream);
      launch_fill(dnm, 0.0f, (long)N4 * 4, stream);
      launch_fill(gout, 0.0f, (long)N4 * 512, stream);
      gat_edge_max_kernel<<<gridFor((long)ET * 4, 256), 256, 0, stream>>>(cei2, E2, N4, a_s, a_d, ebuf, mgat);
      gat_edge_exp_kernel<<<gridFor((long)ET * 4, 256), 256, 0, stream>>>(cei2, E2, N4, ebuf, mgat, dnm);
      gat_edge_aggr_kernel<<<gridFor((long)ET * 4, 256), 256, 0, stream>>>(cei2, E2, N4, ebuf, dnm, hgat, gout);
      gat_finalize_kernel<<<gridFor((long)N4 * 128, 256), 256, 0, stream>>>(gout, gat_b + l * 128, outb, N4);
    }
  }

  // ---- fuse scales via per-node MHA over L=3 ------------------------------
  stack_kernel<<<gridFor((long)N * 128, 256), 256, 0, stream>>>(x0b, x1a, x2a, stck, N);
  launch_gemm(stck, mha_in_w, mha_in_b, nullptr, qkv, N * 3, 384, 128, 0, stream);
  mha_attn_kernel<<<gridFor((long)N * 4, 256), 256, 0, stream>>>(qkv, obar, N);
  // mean over L commutes with the output projection -> one GEMM on obar
  launch_gemm(obar, mha_out_w, mha_out_b, nullptr, comb, N, 128, 128, 0, stream);

  // ---- layernorm + final projection ---------------------------------------
  ln_kernel<<<gridFor((long)N, 256), 256, 0, stream>>>(comb, ln_g, ln_b, nrm, N);
  launch_gemm(nrm, fp_w, fp_b, nullptr, (float*)d_out, N, 128, 128, 0, stream);
}
